// E_GCL_18339510354274
// MI455X (gfx1250) — matrix operand order repaired and checked by reading, not compiled, not run on hardware
//
#include <hip/hip_runtime.h>
#include <hip/hip_bf16.h>

// E_GCL (EGNN layer) fused for MI455X / gfx1250, wave32 + WMMA bf16.
// Round 3: double-buffered weight slabs (stage ks+1 while computing ks),
// one barrier per k-step; everything else as round 2 (b128 fragment loads,
// swizzled weight pool, vectorized gather, sAct aliased onto sA).

#define NN 50000
#define EE 800000
#define FF 128
#define HH 128
#define TILE 64          // edges (or nodes) per block
#define THREADS 256      // 8 wave32 waves

typedef float  v8f   __attribute__((ext_vector_type(8)));
typedef __bf16 v16bf __attribute__((ext_vector_type(16)));
typedef __bf16 v8bf  __attribute__((ext_vector_type(8)));
typedef __bf16 v2bf  __attribute__((ext_vector_type(2)));

// LDS row strides (bf16 elements); multiples of 8 keep b128 loads 16B-aligned.
#define LDA_A 264          // 256-wide gathered A tile (528B rows, bank step 4)
#define LDA_S 136          // 128-wide activation tile (272B rows, bank step 4)
#define BGRP  24           // per-(ntile,lane) group stride in sB: 16 data + 8 pad
#define SLAB  (8 * 32 * BGRP)  // 6144 bf16 = one staged 32x128 weight slab

__device__ __forceinline__ v8f vzero8() {
  v8f v;
#pragma unroll
  for (int i = 0; i < 8; ++i) v[i] = 0.0f;
  return v;
}

__device__ __forceinline__ float silu(float x) {
  return x / (1.0f + __expf(-x));
}

// A fragment (16-bit 16x32): elements 0..15 <- K=half*16+0..15 (contiguous,
// matching the swizzled B pool), row M = lane&15.  Two ds_load_b128.
__device__ __forceinline__ v16bf load_a_frag(const __bf16* sA, int lda, int m0,
                                             int k0, int lane) {
  const int base = (m0 + (lane & 15)) * lda + k0 + ((lane >> 4) << 4);
  const v8bf lo = *(const v8bf*)(sA + base);
  const v8bf hi = *(const v8bf*)(sA + base + 8);
  return __builtin_shufflevector(lo, hi, 0, 1, 2, 3, 4, 5, 6, 7,
                                 8, 9, 10, 11, 12, 13, 14, 15);
}

// B fragment: weight pool is pre-swizzled fragment-major; LDS slab keeps each
// lane's 16 values contiguous (stride BGRP, 2-way bank replication only).
__device__ __forceinline__ v16bf load_b_frag(const __bf16* sB, int nt, int lane) {
  const int g = (nt * 32 + lane) * BGRP;
  const v8bf lo = *(const v8bf*)(sB + g);
  const v8bf hi = *(const v8bf*)(sB + g + 8);
  return __builtin_shufflevector(lo, hi, 0, 1, 2, 3, 4, 5, 6, 7,
                                 8, 9, 10, 11, 12, 13, 14, 15);
}

// Stage one 4096-element swizzled slab into LDS (b128 copies, 2 iters/thread).
__device__ __forceinline__ void stage_slab(__bf16* dst, const __bf16* __restrict__ src,
                                           int tid) {
  const v8bf* s = (const v8bf*)src;
#pragma unroll
  for (int c = tid; c < 512; c += THREADS) {
    const int g = c >> 1, hh = c & 1;
    *(v8bf*)(dst + g * BGRP + hh * 8) = s[c];
  }
}

// LDS-resident GEMM: [TILE x 32*KSTEPS] bf16 (sAbuf) x [32*KSTEPS x 128]
// (swizzled bf16 weights, double-buffered slabs: stage ks+1 during compute ks).
// Wave w: M-tile mt = w&3, N-tiles 4*(w>>2)..+3.  fp32 accumulate.
// Begins with stage+barrier and ends with a barrier -> callers may rely on
// ordering of prior LDS writes and may immediately reuse sAbuf afterwards.
template <int KSTEPS>
__device__ __forceinline__ void gemm_lds(const __bf16* sAbuf, int lda,
                                         const __bf16* __restrict__ gW,
                                         __bf16* sB, int mt, int ngrp,
                                         int lane, int tid, v8f acc[4]) {
  stage_slab(sB, gW, tid);
  __builtin_prefetch(gW + 4096 + tid * 16, 0, 3);  // global_prefetch_b8
  __syncthreads();
  for (int ks = 0; ks < KSTEPS; ++ks) {
    __bf16* cur = sB + (ks & 1) * SLAB;
    __bf16* nxt = sB + ((ks + 1) & 1) * SLAB;
    if (ks + 1 < KSTEPS) {
      stage_slab(nxt, gW + (ks + 1) * 4096, tid);   // overlap with WMMA below
      if (ks + 2 < KSTEPS)
        __builtin_prefetch(gW + (ks + 2) * 4096 + tid * 16, 0, 3);
    }
    const v16bf a = load_a_frag(sAbuf, lda, mt * 16, ks * 32, lane);
#pragma unroll
    for (int t = 0; t < 4; ++t) {
      const v16bf b = load_b_frag(cur, ngrp * 4 + t, lane);
      acc[t] = __builtin_amdgcn_wmma_f32_16x16x32_bf16(
          false, a, false, b, (short)0, acc[t], false, false);
    }
    __syncthreads();  // cur reads done (safe to restage), nxt stores visible
  }
}

// ---------------------------------------------------------------- prep kernels
__global__ __launch_bounds__(256) void egcl_zero_ws(float* wsf, long n) {
  long i = (long)blockIdx.x * blockDim.x + threadIdx.x;
  const long stride = (long)gridDim.x * blockDim.x;
  for (; i < n; i += stride) wsf[i] = 0.0f;
}

// f32 -> bf16 weight conversion with fragment-major swizzle:
// out[((ks*8 + nt)*32 + lane)*16 + e] = W[(ks*32 + (lane>>4)*16 + e)*128 + nt*16 + (lane&15)]
__global__ __launch_bounds__(256) void egcl_cvt_weights(
    const float* __restrict__ We1, const float* __restrict__ We2,
    const float* __restrict__ Wc1, const float* __restrict__ Wn1,
    const float* __restrict__ Wn2, __bf16* __restrict__ pool) {
  const int i = blockIdx.x * blockDim.x + threadIdx.x;
  if (i >= 114688) return;
  const float* W;
  int local, base;
  if (i < 32768)      { W = We1; base = 0;     local = i; }
  else if (i < 49152) { W = We2; base = 32768; local = i - 32768; }
  else if (i < 65536) { W = Wc1; base = 49152; local = i - 49152; }
  else if (i < 98304) { W = Wn1; base = 65536; local = i - 65536; }
  else                { W = Wn2; base = 98304; local = i - 98304; }
  const int e    = local & 15;
  const int lane = (local >> 4) & 31;
  const int nt   = (local >> 9) & 7;
  const int ks   = local >> 12;
  const int k = (ks << 5) + ((lane >> 4) << 4) + e;
  const int n = (nt << 4) + (lane & 15);
  pool[base + local] = (__bf16)W[k * 128 + n];
}

// ---------------------------------------------------------------- edge kernel
__global__ __launch_bounds__(THREADS) void egcl_edge_kernel(
    const float* __restrict__ h, const float* __restrict__ coord,
    const int* __restrict__ ei, const float* __restrict__ We1f,
    const float* __restrict__ be1, const float* __restrict__ be2,
    const float* __restrict__ bc1, const float* __restrict__ Wc2,
    const __bf16* __restrict__ we1b, const __bf16* __restrict__ we2b,
    const __bf16* __restrict__ wc1b,
    float* __restrict__ agg, float* __restrict__ sum_trans,
    float* __restrict__ cnt) {
  __shared__ __bf16 sA[TILE * LDA_A];      // 33792 B; aliased as sAct later
  __shared__ __bf16 sB[2 * SLAB];          // 24576 B (double buffer)
  __shared__ int   sRowI[TILE], sColI[TILE];
  __shared__ float sDiff[TILE * 3], sRad[TILE];
  __shared__ float sW1L[128], sBe1[128], sBe2[128], sBc1[128], sWc2[128];
  __shared__ float sWsum[TILE];
  __bf16* sAct = sA;  // alias: safe, gemm_lds starts and ends with barriers

  const int tid = threadIdx.x;
  const int lane = tid & 31;
  const int w = tid >> 5;
  const int mt = w & 3, ngrp = w >> 2;
  const int half = lane >> 4, nb = lane & 15;
  const int e0 = blockIdx.x * TILE;

  // phase A: indices + per-column constants
  if (tid < TILE) {
    sRowI[tid] = ei[e0 + tid];
    sColI[tid] = ei[EE + e0 + tid];
    sWsum[tid] = 0.0f;
  } else if (tid < TILE + 128) {
    const int j = tid - TILE;
    sW1L[j] = We1f[256 * 128 + j];  // radial row of We1 (rank-1 term, fp32)
    sBe1[j] = be1[j];  sBe2[j] = be2[j];  sBc1[j] = bc1[j];  sWc2[j] = Wc2[j];
  }
  __syncthreads();

  // phase B: geometry + gather e_in = [h[row] | h[col]] as bf16 (float2 loads)
  if (tid < TILE) {
    const int r = sRowI[tid], c = sColI[tid];
    const float dx = coord[r * 3 + 0] - coord[c * 3 + 0];
    const float dy = coord[r * 3 + 1] - coord[c * 3 + 1];
    const float dz = coord[r * 3 + 2] - coord[c * 3 + 2];
    sDiff[tid * 3 + 0] = dx; sDiff[tid * 3 + 1] = dy; sDiff[tid * 3 + 2] = dz;
    sRad[tid] = dx * dx + dy * dy + dz * dz;
  }
#pragma unroll
  for (int idx = tid; idx < TILE * 128; idx += THREADS) {  // pairs of columns
    const int e = idx >> 7, c2 = idx & 127;
    const int node = (c2 < 64) ? sRowI[e] : sColI[e];
    const float2 hv = *(const float2*)(h + (long)node * 128 + ((c2 & 63) << 1));
    v2bf pv;
    pv[0] = (__bf16)hv.x;
    pv[1] = (__bf16)hv.y;
    *(v2bf*)(sA + e * LDA_A + (c2 << 1)) = pv;
  }
  // (gemm_lds' leading barrier orders the gather before any A reads)

  // GEMM 1: m = e_in @ We1  (K=256 via WMMA, +1 radial column via rank-1 VALU)
  v8f acc[4];
#pragma unroll
  for (int t = 0; t < 4; ++t) acc[t] = vzero8();
  gemm_lds<8>(sA, LDA_A, we1b, sB, mt, ngrp, lane, tid, acc);

#pragma unroll
  for (int t = 0; t < 4; ++t) {
    const int n = (ngrp * 4 + t) * 16 + nb;
#pragma unroll
    for (int v = 0; v < 8; ++v) {
      const int m = mt * 16 + half * 8 + v;
      const float x = acc[t][v] + sRad[m] * sW1L[n] + sBe1[n];
      sAct[m * LDA_S + n] = (__bf16)silu(x);
    }
  }
  // gemm_lds(GEMM2) leading barrier orders these sAct writes before A reads

  // GEMM 2: edge_feat = silu(m @ We2 + be2); atomic agg into [N,H]
#pragma unroll
  for (int t = 0; t < 4; ++t) acc[t] = vzero8();
  gemm_lds<4>(sAct, LDA_S, we2b, sB, mt, ngrp, lane, tid, acc);

#pragma unroll
  for (int t = 0; t < 4; ++t) {
    const int n = (ngrp * 4 + t) * 16 + nb;
#pragma unroll
    for (int v = 0; v < 8; ++v) {
      const int m = mt * 16 + half * 8 + v;
      const float x = silu(acc[t][v] + sBe2[n]);
      atomicAdd(&agg[(long)sRowI[m] * 128 + n], x);  // segment_sum(edge_feat)
      sAct[m * LDA_S + n] = (__bf16)x;               // feed coord MLP
    }
  }

  // GEMM 3: p = silu(edge_feat @ Wc1 + bc1); w = p @ Wc2 (dot + lane reduce)
#pragma unroll
  for (int t = 0; t < 4; ++t) acc[t] = vzero8();
  gemm_lds<4>(sAct, LDA_S, wc1b, sB, mt, ngrp, lane, tid, acc);

  float psum[8];
#pragma unroll
  for (int v = 0; v < 8; ++v) psum[v] = 0.0f;
#pragma unroll
  for (int t = 0; t < 4; ++t) {
    const int n = (ngrp * 4 + t) * 16 + nb;
    const float wc = sWc2[n];
#pragma unroll
    for (int v = 0; v < 8; ++v)
      psum[v] += silu(acc[t][v] + sBc1[n]) * wc;
  }
#pragma unroll
  for (int v = 0; v < 8; ++v) {
    float s = psum[v];
    s += __shfl_xor(s, 1, 16);
    s += __shfl_xor(s, 2, 16);
    s += __shfl_xor(s, 4, 16);
    s += __shfl_xor(s, 8, 16);
    if (nb == 0) atomicAdd(&sWsum[mt * 16 + half * 8 + v], s);  // ds_add_f32
  }
  __syncthreads();

  if (tid < TILE) {
    const float wv = sWsum[tid];
    const int r = sRowI[tid];
    atomicAdd(&sum_trans[(long)r * 3 + 0], sDiff[tid * 3 + 0] * wv);
    atomicAdd(&sum_trans[(long)r * 3 + 1], sDiff[tid * 3 + 1] * wv);
    atomicAdd(&sum_trans[(long)r * 3 + 2], sDiff[tid * 3 + 2] * wv);
    atomicAdd(&cnt[r], 1.0f);
  }
}

// ---------------------------------------------------------------- node kernel
__global__ __launch_bounds__(THREADS) void egcl_node_kernel(
    const float* __restrict__ h, const float* __restrict__ coord,
    const float* __restrict__ bn1, const float* __restrict__ bn2,
    const __bf16* __restrict__ wn1b, const __bf16* __restrict__ wn2b,
    const float* __restrict__ agg, const float* __restrict__ sum_trans,
    const float* __restrict__ cnt,
    float* __restrict__ hout, float* __restrict__ coordout) {
  __shared__ __bf16 sA[TILE * LDA_A];
  __shared__ __bf16 sB[2 * SLAB];
  __shared__ float sBn1[128], sBn2[128];
  __bf16* sAct = sA;

  const int tid = threadIdx.x;
  const int lane = tid & 31;
  const int w = tid >> 5;
  const int mt = w & 3, ngrp = w >> 2;
  const int half = lane >> 4, nb = lane & 15;
  const int n0 = blockIdx.x * TILE;

  if (tid < 128) { sBn1[tid] = bn1[tid]; sBn2[tid] = bn2[tid]; }

#pragma unroll
  for (int idx = tid; idx < TILE * 128; idx += THREADS) {
    const int m = idx >> 7, c2 = idx & 127;
    const int node = n0 + m;
    v2bf pv;
    if (node < NN) {
      const float* srcrow = (c2 < 64) ? (h + (long)node * 128)
                                      : (agg + (long)node * 128);
      const float2 hv = *(const float2*)(srcrow + ((c2 & 63) << 1));
      pv[0] = (__bf16)hv.x;
      pv[1] = (__bf16)hv.y;
    } else {
      pv[0] = (__bf16)0.0f;
      pv[1] = (__bf16)0.0f;
    }
    *(v2bf*)(sA + m * LDA_A + (c2 << 1)) = pv;
  }

  v8f acc[4];
#pragma unroll
  for (int t = 0; t < 4; ++t) acc[t] = vzero8();
  gemm_lds<8>(sA, LDA_A, wn1b, sB, mt, ngrp, lane, tid, acc);

#pragma unroll
  for (int t = 0; t < 4; ++t) {
    const int n = (ngrp * 4 + t) * 16 + nb;
#pragma unroll
    for (int v = 0; v < 8; ++v) {
      const int m = mt * 16 + half * 8 + v;
      sAct[m * LDA_S + n] = (__bf16)silu(acc[t][v] + sBn1[n]);
    }
  }

#pragma unroll
  for (int t = 0; t < 4; ++t) acc[t] = vzero8();
  gemm_lds<4>(sAct, LDA_S, wn2b, sB, mt, ngrp, lane, tid, acc);

#pragma unroll
  for (int t = 0; t < 4; ++t) {
    const int n = (ngrp * 4 + t) * 16 + nb;
#pragma unroll
    for (int v = 0; v < 8; ++v) {
      const int m = mt * 16 + half * 8 + v;
      const int node = n0 + m;
      if (node < NN)
        hout[(long)node * 128 + n] =
            h[(long)node * 128 + n] + acc[t][v] + sBn2[n];  // residual
    }
  }

  if (tid < TILE) {
    const int node = n0 + tid;
    if (node < NN) {
      float c = cnt[node];
      c = (c < 1.0f) ? 1.0f : c;
#pragma unroll
      for (int d = 0; d < 3; ++d)
        coordout[(long)node * 3 + d] =
            coord[(long)node * 3 + d] + sum_trans[(long)node * 3 + d] / c;
    }
  }
}

// ---------------------------------------------------------------- launch
extern "C" void kernel_launch(void* const* d_in, const int* in_sizes, int n_in,
                              void* d_out, int out_size, void* d_ws, size_t ws_size,
                              hipStream_t stream) {
  const float* h     = (const float*)d_in[0];
  const float* coord = (const float*)d_in[1];
  const int*   ei    = (const int*)d_in[2];
  const float* We1   = (const float*)d_in[3];
  const float* be1   = (const float*)d_in[4];
  const float* We2   = (const float*)d_in[5];
  const float* be2   = (const float*)d_in[6];
  const float* Wn1   = (const float*)d_in[7];
  const float* bn1   = (const float*)d_in[8];
  const float* Wn2   = (const float*)d_in[9];
  const float* bn2   = (const float*)d_in[10];
  const float* Wc1   = (const float*)d_in[11];
  const float* bc1   = (const float*)d_in[12];
  const float* Wc2   = (const float*)d_in[13];

  float* hout     = (float*)d_out;
  float* coordout = hout + (long)NN * FF;

  // workspace: fp32 accumulators then swizzled bf16 weight pool (32B aligned)
  float* wsf       = (float*)d_ws;
  float* agg       = wsf;                       // N*128
  float* sum_trans = wsf + (long)NN * 128;      // N*3
  float* cnt       = sum_trans + (long)NN * 3;  // N
  __bf16* pool     = (__bf16*)(wsf + (long)NN * 132);
  __bf16* we1b = pool;            // 256x128 swizzled
  __bf16* we2b = pool + 32768;    // 128x128
  __bf16* wc1b = pool + 49152;    // 128x128
  __bf16* wn1b = pool + 65536;    // 256x128
  __bf16* wn2b = pool + 98304;    // 128x128

  const long zcount = (long)NN * 132;
  egcl_zero_ws<<<(int)((zcount + 255) / 256), 256, 0, stream>>>(wsf, zcount);
  egcl_cvt_weights<<<(114688 + 255) / 256, 256, 0, stream>>>(We1, We2, Wc1, Wn1,
                                                             Wn2, pool);

  egcl_edge_kernel<<<EE / TILE, THREADS, 0, stream>>>(
      h, coord, ei, We1, be1, be2, bc1, Wc2, we1b, we2b, wc1b,
      agg, sum_trans, cnt);

  egcl_node_kernel<<<(NN + TILE - 1) / TILE, THREADS, 0, stream>>>(
      h, coord, bn1, bn2, wn1b, wn2b, agg, sum_trans, cnt, hout, coordout);
}
